// DecoderLayer_87574383165921
// MI455X (gfx1250) — matrix operand order repaired and checked by reading, not compiled, not run on hardware
//
#include <hip/hip_runtime.h>

// ---------------- CDNA5 (gfx1250) wave32 WMMA transformer decoder layer ----
// B=2, S=2048, E=1024, H=16, Dh=64, FF=4096.  All GEMMs use
// v_wmma_f32_16x16x32_bf16 (bf16 in, fp32 accum).  GEMM A-tiles are staged
// into LDS with global_load_async_to_lds_b128 (ASYNCcnt).  Attention stages
// shared K/V tiles via the Tensor Data Mover (tensor_load_to_lds, TENSORcnt):
// all 8 waves of a block process the same head, so each 32-key K/V tile is
// DMA'd to LDS once and consumed by all waves.

typedef __attribute__((ext_vector_type(16))) __bf16 v16bf;
typedef __attribute__((ext_vector_type(8)))  float  v8f;
typedef unsigned int v4u __attribute__((ext_vector_type(4)));
typedef unsigned int v8u __attribute__((ext_vector_type(8)));

union BF16Frag { v16bf v; uint4 q[2]; unsigned short s[16]; };

__device__ __forceinline__ unsigned short f2bf(float f) {
  unsigned int u = __float_as_uint(f);
  u += 0x7FFFu + ((u >> 16) & 1u);          // round-to-nearest-even
  return (unsigned short)(u >> 16);
}

__device__ __forceinline__ v8f wmma_bf16(const BF16Frag& a, const BF16Frag& b, v8f c) {
  return __builtin_amdgcn_wmma_f32_16x16x32_bf16(
      false, a.v, false, b.v, (short)0, c, false, false);
}

// Async 16B copy global -> LDS (per-lane), tracked by ASYNCcnt.
// Low 32 bits of a generic pointer into LDS are the raw LDS byte offset
// (ISA 10.2: LDS aperture check uses addr[63:32] only).
__device__ __forceinline__ void async_copy16(unsigned ldsOff, const void* gptr) {
  asm volatile("global_load_async_to_lds_b128 %0, %1, off"
               :: "v"(ldsOff), "v"((unsigned long long)gptr) : "memory");
}
__device__ __forceinline__ void wait_async() {
  asm volatile("s_wait_asynccnt 0x0" ::: "memory");
}

// Tensor Data Mover: 2D tile (tileH rows x tileW bf16) global -> LDS, dense
// in LDS.  D# per ISA 08: group0 = {count|flags, lds_addr, global_addr,
// type=2}; group1 = {data_size=2B, tensor dims == tile dims, row stride}.
// VADDR2/VADDR3 = NULL (<=2D).  Tracked by TENSORcnt.
__device__ __forceinline__ void tdm_load_2d(unsigned ldsOff, const void* gptr,
                                            unsigned tileW, unsigned tileH,
                                            unsigned strideHalves) {
  unsigned long long ga = (unsigned long long)gptr;
  v4u g0;
  g0[0] = 1u;                                        // count=1, user, no gather
  g0[1] = ldsOff;                                    // lds_addr (bytes)
  g0[2] = (unsigned)ga;                              // global_addr[31:0]
  g0[3] = ((unsigned)(ga >> 32) & 0x01FFFFFFu) | (2u << 30);  // [56:32] | type=2
  v8u g1;
  g1[0] = 0x00010000u;                               // data_size=1 (2 bytes)
  g1[1] = (tileW & 0xFFFFu) << 16;                   // tensor_dim0 lo16
  g1[2] = (tileW >> 16) | ((tileH & 0xFFFFu) << 16); // dim0 hi | tensor_dim1 lo
  g1[3] = (tileH >> 16) | (tileW << 16);             // dim1 hi | tile_dim0
  g1[4] = tileH;                                     // tile_dim1 (tile_dim2=0)
  g1[5] = strideHalves;                              // tensor_dim0_stride lo32
  g1[6] = 0u;                                        // stride hi | dim1_stride
  g1[7] = 0u;
  asm volatile("tensor_load_to_lds %0, %1" :: "s"(g0), "s"(g1) : "memory");
}

// ---------------------------------------------------------------- convert --
__global__ __launch_bounds__(256) void f32_to_bf16(const float* __restrict__ src,
                                                   unsigned short* __restrict__ dst,
                                                   int n) {
  for (int i = blockIdx.x * blockDim.x + threadIdx.x; i < n;
       i += gridDim.x * blockDim.x)
    dst[i] = f2bf(src[i]);
}

// --------------------------------------------------------------- layernorm --
__global__ __launch_bounds__(256) void ln_kernel(const float* __restrict__ x,
                                                 const float* __restrict__ g,
                                                 const float* __restrict__ bta,
                                                 unsigned short* __restrict__ out) {
  __shared__ float red[256];
  const int row = blockIdx.x, t = threadIdx.x;
  float4 v = ((const float4*)(x + (size_t)row * 1024))[t];
  red[t] = v.x + v.y + v.z + v.w;
  __syncthreads();
  for (int st = 128; st; st >>= 1) { if (t < st) red[t] += red[t + st]; __syncthreads(); }
  float mu = red[0] * (1.0f / 1024.0f);
  __syncthreads();
  float d0 = v.x - mu, d1 = v.y - mu, d2 = v.z - mu, d3 = v.w - mu;
  red[t] = d0 * d0 + d1 * d1 + d2 * d2 + d3 * d3;
  __syncthreads();
  for (int st = 128; st; st >>= 1) { if (t < st) red[t] += red[t + st]; __syncthreads(); }
  float rstd = rsqrtf(red[0] * (1.0f / 1024.0f) + 1e-5f);
  const int c = t * 4;
  unsigned short* o = out + (size_t)row * 1024 + c;
  o[0] = f2bf(d0 * rstd * g[c + 0] + bta[c + 0]);
  o[1] = f2bf(d1 * rstd * g[c + 1] + bta[c + 1]);
  o[2] = f2bf(d2 * rstd * g[c + 2] + bta[c + 2]);
  o[3] = f2bf(d3 * rstd * g[c + 3] + bta[c + 3]);
}

// -------------------------------------------------------------------- GEMM --
// C[M,N] = epi(A[M,K] @ Bw[K,N] + bias).  Block tile 128x64, 8 waves,
// wave tile 32x32 (4 wmma per 32-K step).  A tile staged via async-to-LDS.
// EPI: 0 = bf16 out, heads layout [B,H,S,Dh]       (Q, K)
//      1 = exact GELU -> bf16 row-major            (FFN1)
//      2 = fp32 out + fp32 residual add            (o-proj, FFN2)
//      3 = bf16 out, transposed heads [B,H,Dh,S]   (V)
template <int EPI>
__global__ __launch_bounds__(256) void gemm_bf16(
    const unsigned short* __restrict__ A, const unsigned short* __restrict__ Bw,
    const float* __restrict__ bias, const float* __restrict__ res,
    unsigned short* __restrict__ outh, float* __restrict__ outf,
    int N, int K) {
  __shared__ unsigned short As[128][32];   // [m][k]
  __shared__ unsigned short Bs[64][32];    // transposed: [n][k]
  const int t = threadIdx.x;
  const int m0 = blockIdx.x * 128, n0 = blockIdx.y * 64;
  const int wid = t >> 5, lane = t & 31;
  const int half = lane >> 4, nn = lane & 15;
  const int wm = (wid & 3) * 32, wn = (wid >> 2) * 32;

  v8f acc[2][2] = {};

  const int ar = t >> 1, ac = (t & 1) * 16;        // A tile: 128 rows x 32 k
  const int bk = t >> 3, bn = (t & 7) * 8;         // B tile: 32 k x 64 n
  const unsigned ldsA = (unsigned)(size_t)(void*)&As[ar][ac];
  const unsigned short* Arow = A + (size_t)(m0 + ar) * K + ac;

  for (int k0 = 0; k0 < K; k0 += 32) {
    async_copy16(ldsA, Arow + k0);                 // A: 16B/lane -> LDS
    union { uint4 q; unsigned short s[8]; } bu;
    bu.q = *(const uint4*)(Bw + (size_t)(k0 + bk) * N + n0 + bn);
#pragma unroll
    for (int i = 0; i < 8; ++i) Bs[bn + i][bk] = bu.s[i];   // transpose into LDS
    wait_async();
    __syncthreads();

    // A frags: lane holds row (lane&15); k = 16*half + e (contiguous)
    BF16Frag a0, a1, b0, b1;
    const unsigned short* ap = &As[wm + nn][0];
    a0.q[0] = *(const uint4*)(ap + 16 * half);
    a0.q[1] = *(const uint4*)(ap + 16 * half + 8);
    ap = &As[wm + 16 + nn][0];
    a1.q[0] = *(const uint4*)(ap + 16 * half);
    a1.q[1] = *(const uint4*)(ap + 16 * half + 8);
    // B frags: lane holds col (lane&15); k = 16*half + e (contiguous in Bs)
    const unsigned short* bp = &Bs[wn + nn][16 * half];
    b0.q[0] = *(const uint4*)bp;       b0.q[1] = *(const uint4*)(bp + 8);
    bp = &Bs[wn + 16 + nn][16 * half];
    b1.q[0] = *(const uint4*)bp;       b1.q[1] = *(const uint4*)(bp + 8);

    acc[0][0] = wmma_bf16(a0, b0, acc[0][0]);
    acc[0][1] = wmma_bf16(a0, b1, acc[0][1]);
    acc[1][0] = wmma_bf16(a1, b0, acc[1][0]);
    acc[1][1] = wmma_bf16(a1, b1, acc[1][1]);
    __syncthreads();
  }

  // Epilogue.  C layout: VGPR r -> row r+8*half, col = lane&15.
#pragma unroll
  for (int i = 0; i < 2; ++i)
#pragma unroll
    for (int j = 0; j < 2; ++j)
#pragma unroll
      for (int r = 0; r < 8; ++r) {
        const int row = m0 + wm + i * 16 + r + 8 * half;
        const int col = n0 + wn + j * 16 + nn;
        float v = acc[i][j][r] + bias[col];
        if (EPI == 0) {  // scatter to [B,H,S,Dh]
          const int b_ = row >> 11, s_ = row & 2047, h_ = col >> 6, d_ = col & 63;
          outh[(((size_t)(b_ * 16 + h_) * 2048 + s_) << 6) + d_] = f2bf(v);
        } else if (EPI == 3) {  // scatter to [B,H,Dh,S]
          const int b_ = row >> 11, s_ = row & 2047, h_ = col >> 6, d_ = col & 63;
          outh[(((size_t)(b_ * 16 + h_) * 64 + d_) << 11) + s_] = f2bf(v);
        } else if (EPI == 1) {
          float gv = 0.5f * v * (1.0f + erff(v * 0.70710678f));
          outh[(size_t)row * N + col] = f2bf(gv);
        } else {
          outf[(size_t)row * N + col] = v + res[(size_t)row * N + col];
        }
      }
}

// ------------------------------------------------- causal streaming attention
// One block = 8 waves = 128 consecutive query rows of ONE head.  K/V tiles
// (32 keys) are TDM-staged into LDS once per block and shared by all waves.
// Scores and P@V via WMMA; unnormalized exp accumulation (scores are O(10)
// here, fp32-safe); one cross-lane denominator reduction at the end.
// Q,K in [B,H,S,Dh]; V in [B,H,Dh,S] (transposed).
__global__ __launch_bounds__(256) void attn_kernel(
    const unsigned short* __restrict__ Q, const unsigned short* __restrict__ Km,
    const unsigned short* __restrict__ Vt, unsigned short* __restrict__ O) {
  __shared__ unsigned short Ks[32][64];      // keys x d      (4 KB)
  __shared__ unsigned short Vs[64][32];      // d x keys      (4 KB)
  __shared__ unsigned short Pb[8][16][32];   // per-wave P    (8 KB)
  const int t = threadIdx.x;
  const int wid = t >> 5, lane = t & 31;
  const int half = lane >> 4, nn = lane & 15;
  const int blk = blockIdx.x;                           // 512 blocks
  const int b_ = blk >> 8, h_ = (blk >> 4) & 15, qs = (blk & 15) * 128;
  const int qbase = qs + wid * 16;                      // this wave's 16 rows
  const int qmax = qbase + 15, kmaxblk = qs + 127;      // block-uniform bound
  const size_t hb = (size_t)(b_ * 16 + h_) * 2048 * 64;
  const unsigned short* Qh = Q + hb;
  const unsigned short* Kh = Km + hb;
  const unsigned short* Vh = Vt + hb;
  const unsigned ldsKs = (unsigned)(size_t)(void*)&Ks[0][0];
  const unsigned ldsVs = (unsigned)(size_t)(void*)&Vs[0][0];

  // Q A-frags (rows qbase..qbase+15, d split 0..31 / 32..63)
  BF16Frag qa0, qa1;
  const unsigned short* qp = Qh + (size_t)(qbase + nn) * 64;
  qa0.q[0] = *(const uint4*)(qp + 16 * half);
  qa0.q[1] = *(const uint4*)(qp + 8 + 16 * half);
  qa1.q[0] = *(const uint4*)(qp + 32 + 16 * half);
  qa1.q[1] = *(const uint4*)(qp + 40 + 16 * half);

  v8f ov[4] = {};
  float lsum[8] = {};

  for (int kb = 0; kb <= kmaxblk; kb += 32) {
    if (wid == 0) {   // wave 0 DMAs the shared K/V tiles (TENSORcnt-tracked)
      tdm_load_2d(ldsKs, Kh + (size_t)kb * 64, /*W*/64, /*H*/32, /*stride*/64);
      tdm_load_2d(ldsVs, Vh + kb,              /*W*/32, /*H*/64, /*stride*/2048);
      __builtin_amdgcn_s_wait_tensorcnt(0);
    }
    __syncthreads();

    if (kb <= qmax) {   // wave-uniform causal skip (EXEC stays all-ones)
      // ---- scores: two 16-wide key tiles, K-dim 64 via two chained WMMA ---
      v8f s0 = {}, s1 = {};
      BF16Frag kf;
      const unsigned short* kp = &Ks[nn][16 * half];
      kf.q[0] = *(const uint4*)kp;        kf.q[1] = *(const uint4*)(kp + 8);
      s0 = wmma_bf16(qa0, kf, s0);
      kf.q[0] = *(const uint4*)(kp + 32); kf.q[1] = *(const uint4*)(kp + 40);
      s0 = wmma_bf16(qa1, kf, s0);
      kp = &Ks[16 + nn][16 * half];
      kf.q[0] = *(const uint4*)kp;        kf.q[1] = *(const uint4*)(kp + 8);
      s1 = wmma_bf16(qa0, kf, s1);
      kf.q[0] = *(const uint4*)(kp + 32); kf.q[1] = *(const uint4*)(kp + 40);
      s1 = wmma_bf16(qa1, kf, s1);

      // ---- exp + causal mask; accumulate per-lane partial denominators ----
#pragma unroll
      for (int r = 0; r < 8; ++r) {
        const int qg = qbase + r + 8 * half;
        float v0 = s0[r] * 0.125f; if (kb + nn > qg)      v0 = -__builtin_inff();
        float v1 = s1[r] * 0.125f; if (kb + 16 + nn > qg) v1 = -__builtin_inff();
        const float e0 = __expf(v0), e1 = __expf(v1);  // masked -> exp(-inf)=0
        lsum[r] += e0 + e1;
        Pb[wid][r + 8 * half][nn]      = f2bf(e0);
        Pb[wid][r + 8 * half][16 + nn] = f2bf(e1);
      }
      // same-wave LDS RAW: DS ops are in-order; force issue-order + data wait
      asm volatile("s_wait_dscnt 0x0" ::: "memory");

      // ---- P (16x32) A-frag from LDS, then O += P @ V (4 N-tiles of d) ----
      BF16Frag pa;
      const unsigned short* pp = &Pb[wid][nn][0];
      pa.q[0] = *(const uint4*)(pp + 16 * half);
      pa.q[1] = *(const uint4*)(pp + 8 + 16 * half);
#pragma unroll
      for (int n = 0; n < 4; ++n) {
        BF16Frag vf;   // B layout: col d = n*16+nn; k = key contiguous in Vs
        const unsigned short* vp = &Vs[n * 16 + nn][16 * half];
        vf.q[0] = *(const uint4*)vp; vf.q[1] = *(const uint4*)(vp + 8);
        ov[n] = wmma_bf16(pa, vf, ov[n]);
      }
    }
    __syncthreads();   // protect Ks/Vs (WAR vs next TDM)
  }

  // ---- one row reduction of denominators, normalize, scatter to [B,S,E] ---
#pragma unroll
  for (int r = 0; r < 8; ++r) {
    float l = lsum[r];
#pragma unroll
    for (int off = 1; off < 16; off <<= 1) l += __shfl_xor(l, off, 32);
    const float inv = 1.0f / l;
    const int row = b_ * 2048 + qbase + r + 8 * half;
#pragma unroll
    for (int n = 0; n < 4; ++n)
      O[(size_t)row * 1024 + h_ * 64 + n * 16 + nn] = f2bf(ov[n][r] * inv);
  }
}

// ------------------------------------------------------------------ launch --
extern "C" void kernel_launch(void* const* d_in, const int* in_sizes, int n_in,
                              void* d_out, int out_size, void* d_ws, size_t ws_size,
                              hipStream_t stream) {
  const float* x    = (const float*)d_in[0];
  // d_in[1] = causal mask (tril) -> implemented analytically in attn_kernel
  const float* ln1g = (const float*)d_in[2];
  const float* ln1b = (const float*)d_in[3];
  const float* wq   = (const float*)d_in[4];
  const float* bq   = (const float*)d_in[5];
  const float* wk   = (const float*)d_in[6];
  const float* bk   = (const float*)d_in[7];
  const float* wv   = (const float*)d_in[8];
  const float* bv   = (const float*)d_in[9];
  const float* wo   = (const float*)d_in[10];
  const float* bo   = (const float*)d_in[11];
  const float* ln2g = (const float*)d_in[12];
  const float* ln2b = (const float*)d_in[13];
  const float* w1   = (const float*)d_in[14];
  const float* b1   = (const float*)d_in[15];
  const float* w2   = (const float*)d_in[16];
  const float* b2   = (const float*)d_in[17];
  (void)in_sizes; (void)n_in; (void)out_size; (void)ws_size;

  unsigned short* w = (unsigned short*)d_ws;
  size_t o = 0;
  auto take = [&](size_t n) { unsigned short* p = w + o; o += n; return p; };
  unsigned short* h1  = take(4194304);   // LN1 out bf16 [4096,1024]; reused as attn-out
  unsigned short* qb  = take(4194304);   // Q [B,H,S,Dh]; reused as h2
  unsigned short* kbm = take(4194304);   // K [B,H,S,Dh]
  unsigned short* vbm = take(4194304);   // V [B,H,Dh,S] (transposed)
  unsigned short* wqb = take(1048576);
  unsigned short* wkb = take(1048576);
  unsigned short* wvb = take(1048576);
  unsigned short* wob = take(1048576);
  unsigned short* w1b = take(4194304);
  unsigned short* w2b = take(4194304);
  unsigned short* ub  = take(16777216);  // GELU(h2@w1+b1) bf16 [4096,4096]
  float* x1 = (float*)(w + o);           // post-attention residual fp32

  f32_to_bf16<<<512, 256, 0, stream>>>(wq, wqb, 1048576);
  f32_to_bf16<<<512, 256, 0, stream>>>(wk, wkb, 1048576);
  f32_to_bf16<<<512, 256, 0, stream>>>(wv, wvb, 1048576);
  f32_to_bf16<<<512, 256, 0, stream>>>(wo, wob, 1048576);
  f32_to_bf16<<<2048, 256, 0, stream>>>(w1, w1b, 4194304);
  f32_to_bf16<<<2048, 256, 0, stream>>>(w2, w2b, 4194304);

  ln_kernel<<<4096, 256, 0, stream>>>(x, ln1g, ln1b, h1);

  dim3 gE(32, 16);   // M=4096/128, N=1024/64
  gemm_bf16<0><<<gE, 256, 0, stream>>>(h1, wqb, bq, nullptr, qb,  nullptr, 1024, 1024);
  gemm_bf16<0><<<gE, 256, 0, stream>>>(h1, wkb, bk, nullptr, kbm, nullptr, 1024, 1024);
  gemm_bf16<3><<<gE, 256, 0, stream>>>(h1, wvb, bv, nullptr, vbm, nullptr, 1024, 1024);

  attn_kernel<<<512, 256, 0, stream>>>(qb, kbm, vbm, /*attn out reuses*/ h1);

  gemm_bf16<2><<<gE, 256, 0, stream>>>(h1, wob, bo, x, nullptr, x1, 1024, 1024);

  unsigned short* h2 = qb;               // reuse (Q dead after attention)
  ln_kernel<<<4096, 256, 0, stream>>>(x1, ln2g, ln2b, h2);

  dim3 gF(32, 64);   // N=4096/64
  gemm_bf16<1><<<gF, 256, 0, stream>>>(h2, w1b, b1, nullptr, ub, nullptr, 4096, 1024);
  gemm_bf16<2><<<gE, 256, 0, stream>>>(ub, w2b, b2, x1, nullptr, (float*)d_out, 1024, 4096);
}
